// MoE_21096879358051
// MI455X (gfx1250) — matrix operand order repaired and checked by reading, not compiled, not run on hardware
//
#include <hip/hip_runtime.h>
#include <hip/hip_bf16.h>

// Problem constants
#define BB 8
#define SS 512
#define DD 512
#define HH 1024
#define EE 8
#define TT (BB * SS)       // 4096 tokens

// LDS strides (bf16 elements) padded so 16 lanes x 16B row-parallel reads are bank-conflict free
#define LDA 520            // x tile rows (512 + 8)
#define LDH 1032           // h tile rows (1024 + 8)

// Workspace layout (bytes)
#define OFF_XBF   0u                      // TT*DD bf16        = 4 MiB
#define OFF_W1T   4194304u                // EE*HH*DD bf16     = 8 MiB
#define OFF_W2T   12582912u               // EE*DD*HH bf16     = 8 MiB
#define OFF_CNT   20971520u               // EE int
#define OFF_LIST  20971648u               // EE*TT int
#define OFF_GATE  21102720u               // EE*TT float

typedef __attribute__((ext_vector_type(16))) __bf16 v16bf;
typedef __attribute__((ext_vector_type(8)))  __bf16 v8bf;
typedef __attribute__((ext_vector_type(4)))  __bf16 v4bf;
typedef __attribute__((ext_vector_type(8)))  float  v8f;

static __device__ __forceinline__ v16bf combine8(v8bf lo, v8bf hi) {
  return __builtin_shufflevector(lo, hi, 0,1,2,3,4,5,6,7,8,9,10,11,12,13,14,15);
}
static __device__ __forceinline__ v8f wmma_bf16(v16bf a, v16bf b, v8f c) {
  // D = A(16x32 bf16) * B(32x16 bf16) + C(16x16 f32)
  return __builtin_amdgcn_wmma_f32_16x16x32_bf16(false, a, false, b, (short)0, c, false, false);
}
static __device__ __forceinline__ v8f zero8() {
  v8f z = {0.f,0.f,0.f,0.f,0.f,0.f,0.f,0.f};
  return z;
}
static __device__ __forceinline__ v16bf loadA16(const __bf16* p) {
  return combine8(*(const v8bf*)(p), *(const v8bf*)(p + 8));
}

// ---------------------------------------------------------------- init
__global__ void moe_zero(int* __restrict__ cnt) {
  if (threadIdx.x < EE) cnt[threadIdx.x] = 0;
}

// ---------------------------------------------------------------- x fp32 -> bf16
__global__ void moe_convert_x(const float4* __restrict__ in, v4bf* __restrict__ out) {
  int i = blockIdx.x * blockDim.x + threadIdx.x;   // over TT*DD/4 elements
  float4 v = in[i];
  v4bf o;
  o[0] = (__bf16)v.x; o[1] = (__bf16)v.y; o[2] = (__bf16)v.z; o[3] = (__bf16)v.w;
  out[i] = o;
}

// ---------------------------------------------------------------- W fp32[r][c] -> bf16[c][r]
__global__ void moe_transpose_bf16(const float* __restrict__ in, __bf16* __restrict__ out,
                                   int rows, int cols) {
  __shared__ float tile[32][33];
  size_t base = (size_t)blockIdx.z * rows * cols;
  int c0 = blockIdx.x * 32, r0 = blockIdx.y * 32;
  #pragma unroll
  for (int j = 0; j < 32; j += 8)
    tile[threadIdx.y + j][threadIdx.x] =
        in[base + (size_t)(r0 + threadIdx.y + j) * cols + c0 + threadIdx.x];
  __syncthreads();
  #pragma unroll
  for (int j = 0; j < 32; j += 8)
    out[base + (size_t)(c0 + threadIdx.y + j) * rows + r0 + threadIdx.x] =
        (__bf16)tile[threadIdx.x][threadIdx.y + j];
}

// ---------------------------------------------------------------- gating: top-2 softmax + dispatch lists + residual
__global__ void moe_gating(const float* __restrict__ x, const int* __restrict__ mask,
                           const float* __restrict__ wg, float* __restrict__ out,
                           int* __restrict__ cnt, int* __restrict__ list,
                           float* __restrict__ gatev) {
  int wave = threadIdx.x >> 5, lane = threadIdx.x & 31;
  int t = blockIdx.x * 8 + wave;                    // one wave per token
  const float* xr = x + (size_t)t * DD;

  float acc[EE];
  #pragma unroll
  for (int e = 0; e < EE; ++e) acc[e] = 0.f;
  for (int d = lane; d < DD; d += 32) {
    float xv = xr[d];
    const float* wr = wg + (size_t)d * EE;          // 8 contiguous floats
    #pragma unroll
    for (int e = 0; e < EE; ++e) acc[e] += xv * wr[e];
  }
  #pragma unroll
  for (int off = 16; off > 0; off >>= 1) {
    #pragma unroll
    for (int e = 0; e < EE; ++e) acc[e] += __shfl_xor(acc[e], off, 32);
  }

  // residual: out = x (also initializes d_out deterministically each call)
  const float4* x4 = (const float4*)xr;
  float4* o4 = (float4*)(out + (size_t)t * DD);
  #pragma unroll
  for (int i = lane; i < DD / 4; i += 32) o4[i] = x4[i];

  if (lane == 0 && mask[t] != 0) {
    // top-2 (ties -> lower index, matching jax.lax.top_k)
    int i0 = 0; float v0 = acc[0];
    #pragma unroll
    for (int e = 1; e < EE; ++e) if (acc[e] > v0) { v0 = acc[e]; i0 = e; }
    int i1 = -1; float v1 = -3.4e38f;
    #pragma unroll
    for (int e = 0; e < EE; ++e) if (e != i0 && acc[e] > v1) { v1 = acc[e]; i1 = e; }
    float e1 = __expf(v1 - v0);                     // v0 >= v1
    float inv = 1.f / (1.f + e1);
    float g0 = inv, g1 = e1 * inv;
    int p0 = atomicAdd(&cnt[i0], 1);
    list[i0 * TT + p0] = t; gatev[i0 * TT + p0] = g0;
    int p1 = atomicAdd(&cnt[i1], 1);
    list[i1 * TT + p1] = t; gatev[i1 * TT + p1] = g1;
  }
}

// ---------------------------------------------------------------- grouped expert MLP (WMMA bf16, f32 acc)
// grid: (ceil(TT/32), EE), block: 512 threads (16 waves). Tile = 32 gathered tokens.
__global__ void __launch_bounds__(512, 4)
moe_expert(const __bf16* __restrict__ xbf,
           const __bf16* __restrict__ w1t,   // [E][H][D] bf16 (N-major for stage 1 B)
           const __bf16* __restrict__ w2t,   // [E][D][H] bf16 (N-major for stage 2 B)
           const float* __restrict__ b1,     // [E][H]
           const float* __restrict__ b2,     // [E][D]
           const int* __restrict__ cnt,
           const int* __restrict__ list,
           const float* __restrict__ gatev,
           float* __restrict__ out) {
  __shared__ __bf16 xsh[32 * LDA];   // gathered x tile, bf16
  __shared__ __bf16 hsh[32 * LDH];   // relu(xW1+b1) tile, bf16

  int e = blockIdx.y;
  int n_e = cnt[e];
  int tbase = blockIdx.x * 32;
  if (tbase >= n_e) return;

  int wave = threadIdx.x >> 5;
  int lane = threadIdx.x & 31;
  int lrow  = lane & 15;             // row (A) / col (B,C,D) index inside 16-tile
  int lhalf = lane >> 4;             // selects K sub-chunk per ISA layouts

  const int*   liste = list  + e * TT;
  const float* gatee = gatev + e * TT;

  // ---- cooperative gather: 32 token rows of bf16 x into LDS
  {
    int row = threadIdx.x >> 4;      // 0..31
    int sub = threadIdx.x & 15;
    int r = tbase + row;
    int rr = (r < n_e) ? r : (n_e - 1);
    int tok = liste[rr];
    const v8bf* src = (const v8bf*)(xbf + (size_t)tok * DD);
    v8bf* dst = (v8bf*)(&xsh[row * LDA]);
    #pragma unroll
    for (int j = sub; j < DD / 8; j += 16) dst[j] = src[j];
  }
  __syncthreads();

  // ---- stage 1: h(32x1024) = relu(x(32x512) @ W1 + b1)
  // wave covers 2 M-subtiles x 4 N-tiles; software-pipelined (double-buffered A and B)
  {
    const __bf16* a0p = &xsh[(0 * 16 + lrow) * LDA + lhalf * 16];
    const __bf16* a1p = &xsh[(1 * 16 + lrow) * LDA + lhalf * 16];
    const __bf16* bp[4];
    #pragma unroll
    for (int tn = 0; tn < 4; ++tn) {
      int c = wave * 64 + tn * 16 + lrow;          // B column for this lane
      bp[tn] = w1t + ((size_t)e * HH + c) * DD + lhalf * 16;
    }
    v8f acc1[2][4];
    #pragma unroll
    for (int ms = 0; ms < 2; ++ms)
      #pragma unroll
      for (int tn = 0; tn < 4; ++tn) acc1[ms][tn] = zero8();

    v16bf a0c = loadA16(a0p);
    v16bf a1c = loadA16(a1p);
    v16bf bc[4];
    #pragma unroll
    for (int tn = 0; tn < 4; ++tn) bc[tn] = *(const v16bf*)(bp[tn]);

    #pragma unroll
    for (int ks = 0; ks < DD; ks += 32) {
      v16bf a0n, a1n, bn[4];
      const bool last = (ks + 32 >= DD);
      if (!last) {                                 // prefetch next k-step (folds at compile time)
        a0n = loadA16(a0p + ks + 32);
        a1n = loadA16(a1p + ks + 32);
        #pragma unroll
        for (int tn = 0; tn < 4; ++tn) bn[tn] = *(const v16bf*)(bp[tn] + ks + 32);
      }
      #pragma unroll
      for (int tn = 0; tn < 4; ++tn) {
        acc1[0][tn] = wmma_bf16(a0c, bc[tn], acc1[0][tn]);
        acc1[1][tn] = wmma_bf16(a1c, bc[tn], acc1[1][tn]);
      }
      if (!last) {
        a0c = a0n; a1c = a1n;
        #pragma unroll
        for (int tn = 0; tn < 4; ++tn) bc[tn] = bn[tn];
      }
    }
    // bias + relu -> bf16 h in LDS. C/D layout: row = lhalf*8+i, col = lrow
    const float* b1e = b1 + (size_t)e * HH;
    #pragma unroll
    for (int ms = 0; ms < 2; ++ms)
      #pragma unroll
      for (int tn = 0; tn < 4; ++tn) {
        int c = wave * 64 + tn * 16 + lrow;
        float bb = b1e[c];
        #pragma unroll
        for (int i = 0; i < 8; ++i) {
          float v = acc1[ms][tn][i] + bb;
          v = v > 0.f ? v : 0.f;
          hsh[(ms * 16 + lhalf * 8 + i) * LDH + c] = (__bf16)v;
        }
      }
  }
  __syncthreads();

  // ---- stage 2: y(32x512) = h(32x1024) @ W2 + b2
  // wave covers 2 M-subtiles x 2 N-tiles; software-pipelined (double-buffered A and B)
  {
    const __bf16* h0p = &hsh[(0 * 16 + lrow) * LDH + lhalf * 16];
    const __bf16* h1p = &hsh[(1 * 16 + lrow) * LDH + lhalf * 16];
    const __bf16* bp[2];
    #pragma unroll
    for (int tn = 0; tn < 2; ++tn) {
      int c = wave * 32 + tn * 16 + lrow;
      bp[tn] = w2t + ((size_t)e * DD + c) * HH + lhalf * 16;
    }
    v8f acc2[2][2];
    #pragma unroll
    for (int ms = 0; ms < 2; ++ms)
      #pragma unroll
      for (int tn = 0; tn < 2; ++tn) acc2[ms][tn] = zero8();

    v16bf a0c = loadA16(h0p);
    v16bf a1c = loadA16(h1p);
    v16bf bc[2];
    #pragma unroll
    for (int tn = 0; tn < 2; ++tn) bc[tn] = *(const v16bf*)(bp[tn]);

    #pragma unroll
    for (int ks = 0; ks < HH; ks += 32) {
      v16bf a0n, a1n, bn[2];
      const bool last = (ks + 32 >= HH);
      if (!last) {
        a0n = loadA16(h0p + ks + 32);
        a1n = loadA16(h1p + ks + 32);
        #pragma unroll
        for (int tn = 0; tn < 2; ++tn) bn[tn] = *(const v16bf*)(bp[tn] + ks + 32);
      }
      #pragma unroll
      for (int tn = 0; tn < 2; ++tn) {
        acc2[0][tn] = wmma_bf16(a0c, bc[tn], acc2[0][tn]);
        acc2[1][tn] = wmma_bf16(a1c, bc[tn], acc2[1][tn]);
      }
      if (!last) {
        a0c = a0n; a1c = a1n;
        #pragma unroll
        for (int tn = 0; tn < 2; ++tn) bc[tn] = bn[tn];
      }
    }
    // bias, gate scale, scatter-accumulate into out (<=2 experts per token)
    const float* b2e = b2 + (size_t)e * DD;
    #pragma unroll
    for (int ms = 0; ms < 2; ++ms)
      #pragma unroll
      for (int tn = 0; tn < 2; ++tn) {
        int c = wave * 32 + tn * 16 + lrow;
        float bb = b2e[c];
        #pragma unroll
        for (int i = 0; i < 8; ++i) {
          int r = tbase + ms * 16 + lhalf * 8 + i;
          bool valid = r < n_e;
          int rr = valid ? r : tbase;
          int tok = liste[rr];
          float g = valid ? gatee[rr] : 0.f;
          float v = (acc2[ms][tn][i] + bb) * g;
          __hip_atomic_fetch_add(out + (size_t)tok * DD + c, v,
                                 __ATOMIC_RELAXED, __HIP_MEMORY_SCOPE_AGENT);
        }
      }
  }
}

// ---------------------------------------------------------------- launcher
extern "C" void kernel_launch(void* const* d_in, const int* in_sizes, int n_in,
                              void* d_out, int out_size, void* d_ws, size_t ws_size,
                              hipStream_t stream) {
  (void)in_sizes; (void)n_in; (void)out_size; (void)ws_size;
  const float* x    = (const float*)d_in[0];
  const int*   mask = (const int*)d_in[1];
  const float* wg   = (const float*)d_in[2];
  const float* W1   = (const float*)d_in[3];
  const float* b1   = (const float*)d_in[4];
  const float* W2   = (const float*)d_in[5];
  const float* b2   = (const float*)d_in[6];
  float* out = (float*)d_out;

  char* ws = (char*)d_ws;
  __bf16* xbf   = (__bf16*)(ws + OFF_XBF);
  __bf16* w1t   = (__bf16*)(ws + OFF_W1T);
  __bf16* w2t   = (__bf16*)(ws + OFF_W2T);
  int*    cnt   = (int*)  (ws + OFF_CNT);
  int*    list  = (int*)  (ws + OFF_LIST);
  float*  gatev = (float*)(ws + OFF_GATE);

  moe_zero<<<1, 32, 0, stream>>>(cnt);

  moe_convert_x<<<(TT * DD / 4) / 256, 256, 0, stream>>>((const float4*)x, (v4bf*)xbf);

  // W1 [E][D][H] -> [E][H][D] bf16 ; W2 [E][H][D] -> [E][D][H] bf16
  moe_transpose_bf16<<<dim3(HH / 32, DD / 32, EE), dim3(32, 8), 0, stream>>>(W1, w1t, DD, HH);
  moe_transpose_bf16<<<dim3(DD / 32, HH / 32, EE), dim3(32, 8), 0, stream>>>(W2, w2t, HH, DD);

  moe_gating<<<TT / 8, 256, 0, stream>>>(x, mask, wg, out, cnt, list, gatev);

  moe_expert<<<dim3(TT / 32, EE), 512, 0, stream>>>(xbf, w1t, w2t, b1, b2,
                                                    cnt, list, gatev, out);
}
